// TopKSparseAutoencoder_33981781246341
// MI455X (gfx1250) — matrix operand order repaired and checked by reading, not compiled, not run on hardware
//
#include <hip/hip_runtime.h>
#include <hip/hip_bf16.h>
#include <stdint.h>

typedef __bf16 bf16_t;
typedef __attribute__((ext_vector_type(16))) __bf16 v16bf;
typedef __attribute__((ext_vector_type(8)))  __bf16 v8bf;
typedef __attribute__((ext_vector_type(8)))  float  v8f;

#define TOK  4096
#define DIM  2048
#define LAT  16384
#define TOPK 32
#define EPSF 1e-5f

// ---------------------------------------------------------------------------
// CDNA5 async global->LDS copy (ASYNCcnt path): 16 bytes per lane, no VGPR
// transit. lds_off is the wave-relative LDS byte offset (low 32 bits of a
// generic pointer to __shared__ per ISA 10.2 aperture mapping).
// ---------------------------------------------------------------------------
__device__ __forceinline__ void async_copy_b128(uint32_t lds_off, const void* gptr) {
  asm volatile("global_load_async_to_lds_b128 %0, %1, off"
               :: "v"(lds_off), "v"(gptr) : "memory");
}
__device__ __forceinline__ void wait_async0() {
  asm volatile("s_wait_asynccnt 0x0" ::: "memory");
}
__device__ __forceinline__ uint32_t lds_off_u32(const void* p) {
  return (uint32_t)(uintptr_t)p;
}

// ---------------------------------------------------------------------------
// 1) Per-token normalization (mean, unbiased std) -> bf16 xn
// ---------------------------------------------------------------------------
__global__ __launch_bounds__(256)
void sae_normalize(const float* __restrict__ x, bf16_t* __restrict__ xnb) {
  __shared__ float rs[256], rq[256];
  const int row = blockIdx.x, t = threadIdx.x;
  const float* xr = x + (size_t)row * DIM;
  float s = 0.f, q = 0.f;
  for (int i = t; i < DIM; i += 256) { float v = xr[i]; s += v; q += v * v; }
  rs[t] = s; rq[t] = q; __syncthreads();
  for (int k = 128; k > 0; k >>= 1) {
    if (t < k) { rs[t] += rs[t + k]; rq[t] += rq[t + k]; }
    __syncthreads();
  }
  const float mean = rs[0] / (float)DIM;
  const float var  = fmaxf((rq[0] - (float)DIM * mean * mean) / (float)(DIM - 1), 0.f);
  const float inv  = 1.f / (sqrtf(var) + EPSF);
  bf16_t* o = xnb + (size_t)row * DIM;
  for (int i = t; i < DIM; i += 256) o[i] = (bf16_t)((xr[i] - mean) * inv);
}

// ---------------------------------------------------------------------------
// 2) W_enc fp32 -> bf16 (one-time convert for the WMMA path)
// ---------------------------------------------------------------------------
__global__ __launch_bounds__(256)
void sae_cvt_w(const float* __restrict__ W, bf16_t* __restrict__ Wb) {
  const size_t base = ((size_t)blockIdx.x * 256 + threadIdx.x) * 8;
  const float4 a = *reinterpret_cast<const float4*>(W + base);
  const float4 b = *reinterpret_cast<const float4*>(W + base + 4);
  v8bf o;
  o[0] = (bf16_t)a.x; o[1] = (bf16_t)a.y; o[2] = (bf16_t)a.z; o[3] = (bf16_t)a.w;
  o[4] = (bf16_t)b.x; o[5] = (bf16_t)b.y; o[6] = (bf16_t)b.z; o[7] = (bf16_t)b.w;
  *reinterpret_cast<v8bf*>(Wb + base) = o;
}

// ---------------------------------------------------------------------------
// 3) Encode GEMM: latents[m,n] = sum_k xn[m,k]*W[n,k] + b_enc[n]
//    Block tile 256(M) x 128(N), 8 waves in 4x2 grid, 64x64 per wave
//    (4x4 WMMA tiles, 128 accumulator VGPRs). K-step 32, async double-buffered
//    LDS staging (ping-pong), one barrier per K-iteration.
// ---------------------------------------------------------------------------
__global__ __launch_bounds__(256)
void sae_encode_gemm(const bf16_t* __restrict__ A,   // [TOK, DIM] bf16
                     const bf16_t* __restrict__ B,   // [LAT, DIM] bf16
                     const float* __restrict__ bias, // [LAT]
                     float* __restrict__ out) {      // [TOK, LAT]
  __shared__ bf16_t sA[2][256][40];  // 40 KB  (+8 bf16 pad keeps 16B align, no bank conflicts)
  __shared__ bf16_t sB[2][128][40];  // 20 KB

  const int t      = threadIdx.x;
  const int lane   = t & 31;
  const int wave   = t >> 5;
  const int lane16 = lane & 15;
  const int hi     = lane >> 4;
  const int wm     = wave >> 1;   // 0..3 -> M offset wm*64
  const int wn     = wave & 1;    // 0..1 -> N offset wn*64
  const int bm     = blockIdx.y * 256;  // token base
  const int bn     = blockIdx.x * 128;  // latent base

  // Staging geometry: per thread 4 A chunks + 2 B chunks of 8 bf16 (16B) each.
  const bf16_t* gA[4]; uint32_t lA[4];
#pragma unroll
  for (int j = 0; j < 4; ++j) {
    const int q = j * 256 + t, row = q >> 2, kp = (q & 3) * 8;
    gA[j] = A + (size_t)(bm + row) * DIM + kp;
    lA[j] = lds_off_u32(&sA[0][row][kp]);
  }
  const bf16_t* gB[2]; uint32_t lB[2];
#pragma unroll
  for (int j = 0; j < 2; ++j) {
    const int q = j * 256 + t, row = q >> 2, kp = (q & 3) * 8;
    gB[j] = B + (size_t)(bn + row) * DIM + kp;
    lB[j] = lds_off_u32(&sB[0][row][kp]);
  }
  const uint32_t bufA = (uint32_t)sizeof(sA[0]);  // 20480
  const uint32_t bufB = (uint32_t)sizeof(sB[0]);  // 10240

  v8f c[4][4];
#pragma unroll
  for (int mi = 0; mi < 4; ++mi)
#pragma unroll
    for (int ni = 0; ni < 4; ++ni) c[mi][ni] = (v8f){};

  // Prologue: stream first K-tile into buffer 0.
#pragma unroll
  for (int j = 0; j < 4; ++j) async_copy_b128(lA[j], gA[j]);
#pragma unroll
  for (int j = 0; j < 2; ++j) async_copy_b128(lB[j], gB[j]);

  int buf = 0;
  for (int kt = 0; kt < DIM; kt += 32, buf ^= 1) {
    wait_async0();        // my ASYNC loads for buffer `buf` have landed
    __syncthreads();      // everyone's landed; previous buffer fully consumed

    if (kt + 32 < DIM) {  // stream next K-tile into the other buffer
      const uint32_t ob = (uint32_t)(buf ^ 1);
#pragma unroll
      for (int j = 0; j < 4; ++j) async_copy_b128(lA[j] + ob * bufA, gA[j] + kt + 32);
#pragma unroll
      for (int j = 0; j < 2; ++j) async_copy_b128(lB[j] + ob * bufB, gB[j] + kt + 32);
    }

    // --- fragments per ISA 7.12.2 lane layouts ---
    union Frag { v16bf v; v8bf h[2]; };
    Frag a[4], bfm[4];
#pragma unroll
    for (int mi = 0; mi < 4; ++mi) {
      const bf16_t* p = &sA[buf][wm * 64 + mi * 16 + lane16][hi * 8];
      a[mi].h[0] = *reinterpret_cast<const v8bf*>(p);       // K 0-7  / 8-15
      a[mi].h[1] = *reinterpret_cast<const v8bf*>(p + 16);  // K 16-23/ 24-31
    }
#pragma unroll
    for (int ni = 0; ni < 4; ++ni) {
      const bf16_t* p = &sB[buf][wn * 64 + ni * 16 + lane16][hi * 8];
      bfm[ni].h[0] = *reinterpret_cast<const v8bf*>(p);     // K 0-7  / 8-15
      bfm[ni].h[1] = *reinterpret_cast<const v8bf*>(p + 16); // K 16-23/ 24-31
    }

#pragma unroll
    for (int mi = 0; mi < 4; ++mi)
#pragma unroll
      for (int ni = 0; ni < 4; ++ni)
        c[mi][ni] = __builtin_amdgcn_wmma_f32_16x16x32_bf16(
            false, a[mi].v, false, bfm[ni].v, (short)0, c[mi][ni], false, false);
  }

  // --- epilogue: C/D layout (VGPR r -> M = r + hi*8, lane16 -> N), fuse +b_enc ---
#pragma unroll
  for (int ni = 0; ni < 4; ++ni) {
    const int n = bn + wn * 64 + ni * 16 + lane16;
    const float bb = bias[n];
#pragma unroll
    for (int mi = 0; mi < 4; ++mi) {
      const int mbase = bm + wm * 64 + mi * 16 + hi * 8;
#pragma unroll
      for (int r = 0; r < 8; ++r)
        out[(size_t)(mbase + r) * LAT + n] = c[mi][ni][r] + bb;
    }
  }
}

// ---------------------------------------------------------------------------
// 4) Top-K=32 per row: LDS bf16 row cache, 32 argmax passes, in-place sparsify
// ---------------------------------------------------------------------------
__global__ __launch_bounds__(256)
void sae_topk(float* __restrict__ lat, int* __restrict__ oidx, float* __restrict__ oval) {
  __shared__ bf16_t srow[LAT];   // 32 KB
  __shared__ float rv[256];
  __shared__ int   ri[256];
  __shared__ int   sidx[TOPK];
  __shared__ float sval[TOPK];
  const int row = blockIdx.x, t = threadIdx.x;
  float* lr = lat + (size_t)row * LAT;

  for (int i = t; i < LAT; i += 256) srow[i] = (bf16_t)lr[i];
  __syncthreads();

  for (int k = 0; k < TOPK; ++k) {
    float best = -3.0e38f; int bi = 0;
    for (int i = t; i < LAT; i += 256) {
      const float v = (float)srow[i];
      if (v > best) { best = v; bi = i; }
    }
    rv[t] = best; ri[t] = bi;
    __syncthreads();
    for (int s = 128; s > 0; s >>= 1) {
      if (t < s && rv[t + s] > rv[t]) { rv[t] = rv[t + s]; ri[t] = ri[t + s]; }
      __syncthreads();
    }
    if (t == 0) {
      sidx[k] = ri[0];
      srow[ri[0]] = (bf16_t)(-__builtin_inff());
    }
    __syncthreads();
  }

  if (t < TOPK) sval[t] = lr[sidx[t]];   // exact fp32 values before zeroing
  __syncthreads();
  for (int i = t; i < LAT; i += 256) lr[i] = 0.f;
  __syncthreads();
  if (t < TOPK) {
    lr[sidx[t]] = sval[t];
    oidx[row * TOPK + t] = sidx[t];
    oval[row * TOPK + t] = sval[t];
  }
}

// ---------------------------------------------------------------------------
// 5) Tied decode: recon[m,:] = pre_bias + sum_j val_j * W_enc[idx_j,:]  (fp32)
// ---------------------------------------------------------------------------
__global__ __launch_bounds__(256)
void sae_decode(const float* __restrict__ W, const float* __restrict__ pre_bias,
                const int* __restrict__ oidx, const float* __restrict__ oval,
                float* __restrict__ recon) {
  __shared__ int   sidx[TOPK];
  __shared__ float sval[TOPK];
  const int row = blockIdx.x, t = threadIdx.x;
  if (t < TOPK) { sidx[t] = oidx[row * TOPK + t]; sval[t] = oval[row * TOPK + t]; }
  __syncthreads();
  float acc[8];
#pragma unroll
  for (int j = 0; j < 8; ++j) acc[j] = pre_bias[t + j * 256];
  for (int k = 0; k < TOPK; ++k) {
    const float* wr = W + (size_t)sidx[k] * DIM;
    const float v = sval[k];
#pragma unroll
    for (int j = 0; j < 8; ++j) acc[j] += v * wr[t + j * 256];
  }
  float* rr = recon + (size_t)row * DIM;
#pragma unroll
  for (int j = 0; j < 8; ++j) rr[t + j * 256] = acc[j];
}

// ---------------------------------------------------------------------------
extern "C" void kernel_launch(void* const* d_in, const int* in_sizes, int n_in,
                              void* d_out, int out_size, void* d_ws, size_t ws_size,
                              hipStream_t stream) {
  const float* x     = (const float*)d_in[0];   // [4096, 2048]
  const float* W_enc = (const float*)d_in[1];   // [16384, 2048]
  const float* b_enc = (const float*)d_in[2];   // [16384]
  const float* pre_b = (const float*)d_in[3];   // [2048]

  float* recon  = (float*)d_out;                          // [4096, 2048]
  float* sparse = (float*)d_out + (size_t)TOK * DIM;      // [4096, 16384]

  char* ws = (char*)d_ws;
  bf16_t* xnb = (bf16_t*)ws;                                         // 16 MB
  bf16_t* Wb  = (bf16_t*)(ws + (size_t)TOK * DIM * 2);               // 64 MB
  int*    oix = (int*)   (ws + (size_t)TOK * DIM * 2 + (size_t)LAT * DIM * 2);
  float*  ovl = (float*) (ws + (size_t)TOK * DIM * 2 + (size_t)LAT * DIM * 2
                             + (size_t)TOK * TOPK * 4);

  sae_normalize<<<TOK, 256, 0, stream>>>(x, xnb);
  sae_cvt_w<<<(int)(((size_t)LAT * DIM) / (256 * 8)), 256, 0, stream>>>(W_enc, Wb);

  dim3 g(LAT / 128, TOK / 256);  // 128 x 16 blocks
  sae_encode_gemm<<<g, 256, 0, stream>>>(xnb, Wb, b_enc, sparse);

  sae_topk<<<TOK, 256, 0, stream>>>(sparse, oix, ovl);
  sae_decode<<<TOK, 256, 0, stream>>>(W_enc, pre_b, oix, ovl, recon);
}
